// MultiLayerE3GNN_77756087926945
// MI455X (gfx1250) — matrix operand order repaired and checked by reading, not compiled, not run on hardware
//
#include <hip/hip_runtime.h>

typedef __attribute__((ext_vector_type(16))) __bf16 v16bf;
typedef __attribute__((ext_vector_type(8)))  float  v8f;

#define N_NODES   2048
#define N_EDGES   49152
#define HID       80
#define MSG_DIM   96
#define WNUM      3328
#define N_LAYERS  4
#define NTILES    208        /* WNUM/16 */
#define NCHUNK    13         /* 208/16  */
#define TILE_E    16

__device__ __forceinline__ float sigmoidf_(float x) { return 1.0f / (1.0f + __expf(-x)); }
__device__ __forceinline__ float siluf_(float x)    { return x * sigmoidf_(x); }

#define S2        0.70710678f   /* 1/sqrt(2)  */
#define S6        0.40824829f   /* 1/sqrt(6)  */
#define C53       0.77459667f   /* sqrt(3/5)  */
#define INV_SQRT3 0.57735027f
#define INV_SQRT48 0.14433757f
#define INV_SQRT24 0.20412415f
#define INV_SQRT32 0.17677670f

// ---------------------------------------------------------------- geometry
__global__ void k_geom(const int* __restrict__ ei, const float* __restrict__ pos,
                       float* __restrict__ ef) {
    int e = blockIdx.x * blockDim.x + threadIdx.x;
    if (e >= N_EDGES) return;
    int s = ei[e], d = ei[N_EDGES + e];
    float rx = pos[d*3+0] - pos[s*3+0];
    float ry = pos[d*3+1] - pos[s*3+1];
    float rz = pos[d*3+2] - pos[s*3+2];
    float r = sqrtf(rx*rx + ry*ry + rz*rz);
    r = fmaxf(r, 1e-6f);
    float inv = 1.0f / r;
    float x = rx*inv, y = ry*inv, z = rz*inv;
    float* o = ef + (size_t)e * 16;
    o[0] = 1.7320508f * x;  o[1] = 1.7320508f * y;  o[2] = 1.7320508f * z;
    const float c15 = 3.8729833f;
    o[3] = c15 * x * y;
    o[4] = c15 * y * z;
    o[5] = 1.1180340f * (3.0f * z * z - 1.0f);
    o[6] = c15 * x * z;
    o[7] = 0.5f * c15 * (x * x - y * y);
    float u = r * (1.0f / 6.0f);
    float u2 = u*u, u3 = u2*u, u6 = u3*u3, u7 = u6*u, u8 = u7*u;
    float fc = (u < 1.0f) ? (1.0f - 28.0f*u6 + 48.0f*u7 - 21.0f*u8) : 0.0f;
    float pref = 0.57735027f * fc * inv;  /* sqrt(2/6) * fc / r */
    #pragma unroll
    for (int n = 1; n <= 8; ++n)
        o[7 + n] = pref * __sinf(0.52359878f * (float)n * r);
}

// ---------------------------------------------------------------- embed
__global__ void k_embed(const float* __restrict__ x, const float* __restrict__ ew,
                        float* __restrict__ h) {
    int gid = blockIdx.x * blockDim.x + threadIdx.x;
    if (gid >= N_NODES * HID) return;
    int n = gid / HID, c = gid % HID;
    float v = 0.0f;
    if (c < 32) {
        #pragma unroll
        for (int k = 0; k < 16; ++k) v += x[n*16 + k] * ew[k*32 + c];
        v *= 0.25f;   /* 1/sqrt(16) */
    }
    h[gid] = v;
}

// -------------------------------------------------- repack W3 + bias -> bf16
// Layout: [layer][ct*3+s][lane(32)][j(16)] = per-lane B-operand order for
// v_wmma_f32_16x16x32_bf16.  Logical repacked columns:
// blk1 w*32+u | blk2 w*16+u | blk3 w*32+u | blk4,5 w*16+u
__global__ void k_repack(const float* __restrict__ w3, const float* __restrict__ b3,
                         __bf16* __restrict__ w3r) {
    int gid = blockIdx.x * blockDim.x + threadIdx.x;
    const int per_layer = NTILES * 3 * 512;
    if (gid >= N_LAYERS * per_layer) return;
    int l = gid / per_layer;
    int o = gid % per_layer;
    int blk = o >> 9;  int ct = blk / 3, s = blk % 3;
    int r = o & 511;   int ln = r >> 4, j = r & 15;
    int k = 32*s + ((ln >> 4) << 3) + (j & 7) + ((j >> 3) << 4);
    int n = ln & 15;
    int c = ct * 16 + n;
    int orig;
    if      (c < 1536) { int w = c >> 5,          u = c & 31;          orig = u*48 + w; }
    else if (c < 2304) { int w = (c-1536) >> 4,   u = (c-1536) & 15;   orig = 1536 + u*48 + w; }
    else if (c < 2816) { int w = (c-2304) >> 5,   u = (c-2304) & 31;   orig = 2304 + u*16 + w; }
    else if (c < 3072) { int w = (c-2816) >> 4,   u = (c-2816) & 15;   orig = 2816 + u*16 + w; }
    else               { int w = (c-3072) >> 4,   u = (c-3072) & 15;   orig = 3072 + u*16 + w; }
    float val;
    if (k < 64)       val = w3[(size_t)l*64*WNUM + (size_t)k*WNUM + orig];
    else if (k == 64) val = b3[(size_t)l*WNUM + orig];
    else              val = 0.0f;
    w3r[gid] = (__bf16)val;
}

// ---------------------------------------------------------------- node linears
__global__ __launch_bounds__(160)
void k_lin(const float* __restrict__ h,
           const float* __restrict__ lws, const float* __restrict__ lwv,
           const float* __restrict__ sws, const float* __restrict__ swv,
           float* __restrict__ hx, float* __restrict__ xself) {
    __shared__ float hl[HID];
    int n = blockIdx.x, t = threadIdx.x;
    if (t < HID) hl[t] = h[(size_t)n*HID + t];
    __syncthreads();
    bool self = (t >= 80);
    int tt = self ? t - 80 : t;
    const float* Ws = self ? sws : lws;
    const float* Wv = self ? swv : lwv;
    float* out = self ? xself : hx;
    if (tt < 32) {
        float a = 0.0f;
        #pragma unroll
        for (int u = 0; u < 32; ++u) a += hl[u] * Ws[u*32 + tt];
        out[(size_t)n*HID + tt] = a * INV_SQRT32;
    } else {
        int idx = tt - 32, w = idx / 3, i = idx % 3;
        float a = 0.0f;
        #pragma unroll
        for (int u = 0; u < 16; ++u) a += hl[32 + u*3 + i] * Wv[u*16 + w];
        out[(size_t)n*HID + 32 + w*3 + i] = a * 0.25f;  /* 1/sqrt(16) */
    }
}

// ---------------------------------------------------------------- radial MLP 1-2
__global__ __launch_bounds__(256)
void k_radial(const float* __restrict__ ef,
              const float* __restrict__ W1, const float* __restrict__ b1,
              const float* __restrict__ W2, const float* __restrict__ b2,
              __bf16* __restrict__ f2b) {
    __shared__ float sW1[8*64], sb1[64], sW2[64*64], sb2[64];
    int t = threadIdx.x;
    for (int i = t; i < 512;  i += 256) sW1[i] = W1[i];
    for (int i = t; i < 4096; i += 256) sW2[i] = W2[i];
    if (t < 64) { sb1[t] = b1[t]; sb2[t] = b2[t]; }
    __syncthreads();
    int e = blockIdx.x * 256 + t;
    float rbf[8];
    #pragma unroll
    for (int k = 0; k < 8; ++k) rbf[k] = ef[(size_t)e*16 + 8 + k];
    float f1[64];
    #pragma unroll
    for (int j = 0; j < 64; ++j) {
        float a = sb1[j];
        #pragma unroll
        for (int k = 0; k < 8; ++k) a += rbf[k] * sW1[k*64 + j];
        f1[j] = siluf_(a);
    }
    #pragma unroll 4
    for (int j = 0; j < 64; ++j) {
        float a = sb2[j];
        #pragma unroll
        for (int k = 0; k < 64; ++k) a += f1[k] * sW2[k*64 + j];
        f2b[(size_t)e*64 + j] = (__bf16)siluf_(a);
    }
}

__global__ void k_zero(float* __restrict__ p, int n) {
    int gid = blockIdx.x * blockDim.x + threadIdx.x;
    if (gid < n) p[gid] = 0.0f;
}

// ---------------------------------------------------------------- fused edge kernel
__global__ __launch_bounds__(256)
void k_edge(const int* __restrict__ ei, const float* __restrict__ hx,
            const float* __restrict__ ef, const __bf16* __restrict__ f2b,
            const __bf16* __restrict__ w3r, float* __restrict__ agg) {
    __shared__ float s_lds[TILE_E][33];
    __shared__ float v_lds[TILE_E][49];
    __shared__ float p2_lds[TILE_E][17];
    __shared__ float q5_lds[TILE_E][49];
    __shared__ float y1_lds[TILE_E][3];
    __shared__ float Y_lds[TILE_E][9];
    __shared__ float msg_lds[TILE_E][MSG_DIM];
    __shared__ float w_lds[2][TILE_E][260];   // double-buffered weight chunk
    __shared__ v16bf a_ldsv[3*32];
    __shared__ int   s_dst[TILE_E];

    const int t = threadIdx.x;
    const int e0 = blockIdx.x * TILE_E;
    const int lane = t & 31, wave = t >> 5;

    // per-edge spherical harmonics + Y matrix
    if (t < TILE_E) {
        int e = t;
        s_dst[e] = ei[N_EDGES + e0 + e];
        const float* o = ef + (size_t)(e0 + e) * 16;
        y1_lds[e][0] = o[0]; y1_lds[e][1] = o[1]; y1_lds[e][2] = o[2];
        float y20 = o[3], y21 = o[4], y22 = o[5], y23 = o[6], y24 = o[7];
        Y_lds[e][0] = -S6*y22 + S2*y24;  Y_lds[e][1] = S2*y20;            Y_lds[e][2] = S2*y23;
        Y_lds[e][3] =  S2*y20;           Y_lds[e][4] = -S6*y22 - S2*y24;  Y_lds[e][5] = S2*y21;
        Y_lds[e][6] =  S2*y23;           Y_lds[e][7] = S2*y21;            Y_lds[e][8] = 2.0f*S6*y22;
    }
    // gather sender features
    {
        int e = t >> 4, c0 = t & 15;
        int sidx = ei[e0 + e];
        for (int c = c0; c < HID; c += 16) {
            float v = hx[(size_t)sidx*HID + c];
            if (c < 32) s_lds[e][c] = v;
            else        v_lds[e][c - 32] = v;
        }
    }
    for (int i = t; i < TILE_E * MSG_DIM; i += 256) ((float*)msg_lds)[i] = 0.0f;
    // build bf16 A-operand (f2 | bias row 1 | zero pad), pre-swizzled per lane
    {
        __bf16* ab = (__bf16*)a_ldsv;
        for (int i = t; i < 3*512; i += 256) {
            int s = i >> 9, r = i & 511, ln = r >> 4, j = r & 15;
            int m = ln & 15, half = ln >> 4;
            int k = (j & 7) + (half << 3) + ((j >> 3) << 4) + (s << 5);
            float val = (k < 64) ? (float)f2b[(size_t)(e0 + m)*64 + k]
                                 : (k == 64 ? 1.0f : 0.0f);
            ab[i] = (__bf16)val;
        }
    }
    __syncthreads();
    // hoist the loop-invariant A operands into registers (one LDS read each)
    const v16bf a0 = a_ldsv[0*32 + lane];
    const v16bf a1 = a_ldsv[1*32 + lane];
    const v16bf a2 = a_ldsv[2*32 + lane];
    // p2 and q5
    {
        int e = t >> 4, u = t & 15;
        float v0 = v_lds[e][u*3+0], v1 = v_lds[e][u*3+1], v2 = v_lds[e][u*3+2];
        p2_lds[e][u] = (v0*y1_lds[e][0] + v1*y1_lds[e][1] + v2*y1_lds[e][2]) * INV_SQRT3;
        #pragma unroll
        for (int j = 0; j < 3; ++j)
            q5_lds[e][u*3+j] = C53 * (v0*Y_lds[e][0*3+j] + v1*Y_lds[e][1*3+j] + v2*Y_lds[e][2*3+j]);
    }
    __syncthreads();

    const v16bf* bmat = (const v16bf*)w3r;
    for (int chunk = 0; chunk < NCHUNK; ++chunk) {
        const int buf = chunk & 1;
        // --- generate 16x256 chunk of per-edge weights with WMMA ---
        #pragma unroll
        for (int t2 = 0; t2 < 2; ++t2) {
            int ct = chunk*16 + wave*2 + t2;
            const v16bf* bp = bmat + (size_t)ct*3*32 + lane;
            v8f acc = {};
            acc = __builtin_amdgcn_wmma_f32_16x16x32_bf16(false, a0, false, bp[0],
                                                          (short)0, acc, false, false);
            acc = __builtin_amdgcn_wmma_f32_16x16x32_bf16(false, a1, false, bp[32],
                                                          (short)0, acc, false, false);
            acc = __builtin_amdgcn_wmma_f32_16x16x32_bf16(false, a2, false, bp[64],
                                                          (short)0, acc, false, false);
            int nc = lane & 15, mb = (lane >> 4) * 8;
            int lc = (wave*2 + t2)*16 + nc;
            #pragma unroll
            for (int p = 0; p < 8; ++p) w_lds[buf][mb + p][lc] = acc[p];
        }
        __syncthreads();   // single barrier: buffer(buf) now complete
        // --- contract chunk into message accumulator (race-free by layout) ---
        if (chunk < 6) {                       // block1: s_j . w1 -> out_s
            if (t < 128) {
                int e = t >> 3, gi = t & 7, w = chunk*8 + gi;
                float a = 0.0f;
                #pragma unroll
                for (int u = 0; u < 32; ++u) a += s_lds[e][u] * w_lds[buf][e][gi*32 + u];
                msg_lds[e][w] += a * INV_SQRT48;
            }
        } else if (chunk < 9) {                // block2: p2 . w2 -> out_s
            int e = t >> 4, gi = t & 15, w = (chunk - 6)*16 + gi;
            float a = 0.0f;
            #pragma unroll
            for (int u = 0; u < 16; ++u) a += p2_lds[e][u] * w_lds[buf][e][gi*16 + u];
            msg_lds[e][w] += a * INV_SQRT48;
        } else if (chunk < 11) {               // block3: y1 (x) (s_j . w3) -> out_v
            if (t < 128) {
                int e = t >> 3, gi = t & 7, w = (chunk - 9)*8 + gi;
                float a = 0.0f;
                #pragma unroll
                for (int u = 0; u < 32; ++u) a += s_lds[e][u] * w_lds[buf][e][gi*32 + u];
                a *= 0.125f;                   /* 1/sqrt(64) */
                #pragma unroll
                for (int i = 0; i < 3; ++i) msg_lds[e][48 + w*3 + i] += y1_lds[e][i] * a;
            }
        } else {                               // block4 (v_j) / block5 (q5) -> out_v
            int e = t >> 4, gi = t & 15;
            const float* F = (chunk == 11) ? &v_lds[e][0] : &q5_lds[e][0];
            float a0c = 0.0f, a1c = 0.0f, a2c = 0.0f;
            #pragma unroll
            for (int u = 0; u < 16; ++u) {
                float wv = w_lds[buf][e][gi*16 + u];
                a0c += F[u*3+0] * wv; a1c += F[u*3+1] * wv; a2c += F[u*3+2] * wv;
            }
            msg_lds[e][48 + gi*3 + 0] += a0c * 0.125f;
            msg_lds[e][48 + gi*3 + 1] += a1c * 0.125f;
            msg_lds[e][48 + gi*3 + 2] += a2c * 0.125f;
        }
        // no trailing barrier: next iteration writes the OTHER w_lds buffer;
        // two-iteration reuse distance is covered by the per-iteration barrier.
    }
    __syncthreads();
    // scatter-add to receiver nodes
    {
        int e = t >> 4, c0 = t & 15;
        int d = s_dst[e];
        for (int c = c0; c < MSG_DIM; c += 16)
            atomicAdd(&agg[(size_t)d*MSG_DIM + c], msg_lds[e][c]);
    }
}

// ---------------------------------------------------------------- node update
__global__ void k_update(float* __restrict__ h, const float* __restrict__ xself,
                         const float* __restrict__ agg) {
    int gid = blockIdx.x * blockDim.x + threadIdx.x;
    if (gid >= N_NODES * HID) return;
    int n = gid / HID, c = gid % HID;
    float res = h[gid] + xself[gid];
    if (c < 32) {
        res += siluf_(agg[(size_t)n*MSG_DIM + c] * INV_SQRT24);
    } else {
        int u = (c - 32) / 3, i = (c - 32) % 3;
        float g   = sigmoidf_(agg[(size_t)n*MSG_DIM + 32 + u] * INV_SQRT24);
        float vec = agg[(size_t)n*MSG_DIM + 48 + u*3 + i] * INV_SQRT24;
        res += vec * g;
    }
    h[gid] = res;
}

// ---------------------------------------------------------------- final projection
__global__ void k_proj(const float* __restrict__ h, const float* __restrict__ pw,
                       float* __restrict__ out) {
    int gid = blockIdx.x * blockDim.x + threadIdx.x;
    if (gid >= N_NODES * 32) return;
    int n = gid / 32, c = gid % 32;
    float a = 0.0f;
    #pragma unroll
    for (int u = 0; u < 32; ++u) a += h[(size_t)n*HID + u] * pw[u*32 + c];
    out[gid] = a * INV_SQRT32;
}

// ================================================================ host launch
extern "C" void kernel_launch(void* const* d_in, const int* in_sizes, int n_in,
                              void* d_out, int out_size, void* d_ws, size_t ws_size,
                              hipStream_t stream) {
    (void)in_sizes; (void)n_in; (void)out_size; (void)ws_size;
    const float* x       = (const float*)d_in[0];
    const float* pos     = (const float*)d_in[1];
    const int*   ei      = (const int*)  d_in[2];
    const float* embed_w = (const float*)d_in[3];
    const float* proj_w  = (const float*)d_in[4];
    const float* lin_ws  = (const float*)d_in[5];
    const float* lin_wv  = (const float*)d_in[6];
    const float* sc_ws   = (const float*)d_in[7];
    const float* sc_wv   = (const float*)d_in[8];
    const float* mlp_w1  = (const float*)d_in[9];
    const float* mlp_b1  = (const float*)d_in[10];
    const float* mlp_w2  = (const float*)d_in[11];
    const float* mlp_b2  = (const float*)d_in[12];
    const float* mlp_w3  = (const float*)d_in[13];
    const float* mlp_b3  = (const float*)d_in[14];
    float* out = (float*)d_out;

    char* ws = (char*)d_ws;
    size_t off = 0;
    auto alloc = [&](size_t bytes) -> void* {
        void* p = ws + off;
        off = (off + bytes + 255) & ~(size_t)255;
        return p;
    };
    float*  ef    = (float*) alloc((size_t)N_EDGES * 16 * 4);
    __bf16* f2b   = (__bf16*)alloc((size_t)N_EDGES * 64 * 2);
    __bf16* w3r   = (__bf16*)alloc((size_t)N_LAYERS * NTILES * 3 * 512 * 2);
    float*  h     = (float*) alloc((size_t)N_NODES * HID * 4);
    float*  hx    = (float*) alloc((size_t)N_NODES * HID * 4);
    float*  xself = (float*) alloc((size_t)N_NODES * HID * 4);
    float*  agg   = (float*) alloc((size_t)N_NODES * MSG_DIM * 4);

    k_geom<<<N_EDGES/256, 256, 0, stream>>>(ei, pos, ef);
    k_embed<<<(N_NODES*HID + 255)/256, 256, 0, stream>>>(x, embed_w, h);
    k_repack<<<(N_LAYERS*NTILES*3*512 + 255)/256, 256, 0, stream>>>(mlp_w3, mlp_b3, w3r);

    for (int l = 0; l < N_LAYERS; ++l) {
        k_lin<<<N_NODES, 160, 0, stream>>>(h,
            lin_ws + (size_t)l*32*32, lin_wv + (size_t)l*16*16,
            sc_ws  + (size_t)l*32*32, sc_wv  + (size_t)l*16*16, hx, xself);
        k_radial<<<N_EDGES/256, 256, 0, stream>>>(ef,
            mlp_w1 + (size_t)l*8*64,  mlp_b1 + (size_t)l*64,
            mlp_w2 + (size_t)l*64*64, mlp_b2 + (size_t)l*64, f2b);
        k_zero<<<(N_NODES*MSG_DIM + 255)/256, 256, 0, stream>>>(agg, N_NODES*MSG_DIM);
        k_edge<<<N_EDGES/TILE_E, 256, 0, stream>>>(ei, hx, ef, f2b,
            w3r + (size_t)l*NTILES*3*512, agg);
        k_update<<<(N_NODES*HID + 255)/256, 256, 0, stream>>>(h, xself, agg);
    }
    k_proj<<<(N_NODES*32 + 255)/256, 256, 0, stream>>>(h, proj_w, out);
}
